// StableQGFDLayer_40681930227828
// MI455X (gfx1250) — matrix operand order repaired and checked by reading, not compiled, not run on hardware
//
#include <hip/hip_runtime.h>

// ---------------------------------------------------------------------------
// StableQGFD layer for MI455X (gfx1250): f16 WMMA GEMMs, f32 accumulate.
// N=2048 tokens, D=1024, H=16 heads, hd=64, 2 diffusion steps, alpha=0.05.
// All GEMMs consume B in transposed storage Bt[n][k] -> contiguous 32B/lane
// fragment loads (global_load_b128 pairs), no scalar gathers.
// ---------------------------------------------------------------------------

typedef _Float16 f16;
typedef __attribute__((ext_vector_type(16))) _Float16 v16h;
typedef __attribute__((ext_vector_type(8)))  _Float16 v8h;
typedef __attribute__((ext_vector_type(8)))  float    v8f;

#define TOKENS   2048
#define EMBED    1024
#define NHEADS   16
#define HEADDIM  64
#define ALPHA_C  0.05f

union FragU { v16h v; v8h h8[2]; };

// --- WMMA wrapper: D = A(16x32 f16) * B(32x16 f16) + C(16x16 f32) ----------
__device__ __forceinline__ v8f wmma_f16(v16h a, v16h b, v8f c) {
  return __builtin_amdgcn_wmma_f32_16x16x32_f16(
      /*neg_a=*/false, a, /*neg_b=*/false, b,
      /*c_mod=*/(short)0, c, /*reuse_a=*/false, /*reuse_b=*/false);
}

// --- A fragment, 16x32 (MxK) f16, row-major A[lda], contiguous k order ----
// lane half h: elem i -> K = kk + 16h + i  (contiguous, matches B)
__device__ __forceinline__ v16h load_a_frag(const f16* __restrict__ A, int lda,
                                            int m, int kk, int h) {
  const f16* base = A + (size_t)m * lda + kk;
  FragU u;
  u.h8[0] = *(const v8h*)(base + h * 16);
  u.h8[1] = *(const v8h*)(base + h * 16 + 8);
  return u.v;
}

// --- B fragment, 32x16 (KxN), from transposed storage Bt[n][k] -------------
// lane half h: elem i -> K = kk + 16h + i  (one contiguous 32-byte run)
__device__ __forceinline__ v16h load_bt_frag(const f16* __restrict__ Bt, int ldb,
                                             int n, int kk, int h) {
  return *(const v16h*)(Bt + (size_t)n * ldb + kk + h * 16);
}

// ---------------------------------------------------------------------------
// WMMA GEMM: C = scale * (A @ Bt^T) [+ beta*blendH] [+ bias]
//   256 threads = 8 waves (4 along M x 2 along N).
//   Wave tile 32 x (16*NSUB); macro tile 128 x (32*NSUB).
//   NSUB=4 -> 128x128 macro tile, 8 wmma / K-step / wave.
//   OUTLAYOUT: 0 = row-major [m][n]
//              1 = head-split   [n/64][m][n%64]   (Q/K projections)
//              2 = head-split-T [n/64][n%64][m]   (V projection -> V^T)
// All dims are exact tile multiples -> no divergence; EXEC all-ones at wmma.
// ---------------------------------------------------------------------------
template <int NSUB, bool BIAS, bool BLEND, bool OF32, bool OF16, int OUTLAYOUT>
__global__ __launch_bounds__(256) void gemm_wmma_kernel(
    const f16* __restrict__ A, int lda,
    const f16* __restrict__ Bt, int ldb,
    const float* __restrict__ bias,
    const f16* __restrict__ blendH, int ldbl,
    float scale, float beta,
    float* __restrict__ Cf, int ldcf,
    f16* __restrict__ Ch, int ldch,
    int K, int tokens) {
  const int lane = (int)threadIdx.x & 31;
  const int wave = (int)threadIdx.x >> 5;
  const int h    = lane >> 4;   // half-wave select
  const int l16  = lane & 15;
  const int wm = wave >> 1;     // 4 waves along M
  const int wn = wave & 1;      // 2 waves along N
  const int m0 = (int)blockIdx.x * 128 + wm * 32;
  const int n0 = (int)blockIdx.y * (32 * NSUB) + wn * (16 * NSUB);

  v8f acc[2][NSUB];
#pragma unroll
  for (int i = 0; i < 2; ++i)
#pragma unroll
    for (int j = 0; j < NSUB; ++j) acc[i][j] = (v8f){};

  for (int kk = 0; kk < K; kk += 32) {
    // hint next K-slab of A into cache (global_prefetch_b8)
    __builtin_prefetch(A + (size_t)(m0 + l16) * lda + kk + 32, 0, 1);

    v16h a[2];
    a[0] = load_a_frag(A, lda, m0 + l16, kk, h);
    a[1] = load_a_frag(A, lda, m0 + 16 + l16, kk, h);
    v16h b[NSUB];
#pragma unroll
    for (int j = 0; j < NSUB; ++j)
      b[j] = load_bt_frag(Bt, ldb, n0 + j * 16 + l16, kk, h);

#pragma unroll
    for (int i = 0; i < 2; ++i)
#pragma unroll
      for (int j = 0; j < NSUB; ++j)
        acc[i][j] = wmma_f16(a[i], b[j], acc[i][j]);
  }

  // Epilogue: C/D layout -> lane half h, VGPR r holds (m = 8h + r, n = l16).
#pragma unroll
  for (int i = 0; i < 2; ++i) {
#pragma unroll
    for (int j = 0; j < NSUB; ++j) {
      const int n = n0 + j * 16 + l16;
#pragma unroll
      for (int r = 0; r < 8; ++r) {
        const int m = m0 + i * 16 + h * 8 + r;
        float v = scale * acc[i][j][r];
        if constexpr (BLEND) v += beta * (float)blendH[(size_t)m * ldbl + n];
        if constexpr (BIAS)  v += bias[n];
        if constexpr (OF32)  Cf[(size_t)m * ldcf + n] = v;
        if constexpr (OF16) {
          if constexpr (OUTLAYOUT == 1)
            Ch[((size_t)(n >> 6) * tokens + m) * HEADDIM + (n & 63)] = (f16)v;
          else if constexpr (OUTLAYOUT == 2)
            Ch[((size_t)(n >> 6) * HEADDIM + (n & 63)) * tokens + m] = (f16)v;
          else
            Ch[(size_t)m * ldch + n] = (f16)v;
        }
      }
    }
  }
}

// --- row softmax: one 256-thread block per row; f16 output with strides ----
// (orow=cols, ocol=1) -> row-major;  (orow=1, ocol=rows) -> transposed.
__global__ __launch_bounds__(256) void softmax_rows_kernel(
    const float* __restrict__ S, f16* __restrict__ outH,
    int cols, int orow, int ocol) {
  const int row = (int)blockIdx.x;
  const int tid = (int)threadIdx.x;
  const float* r = S + (size_t)row * cols;
  __shared__ float red[8];

  float mx = -3.4e38f;
  for (int c = tid; c < cols; c += 256) mx = fmaxf(mx, r[c]);
  for (int o = 16; o > 0; o >>= 1) mx = fmaxf(mx, __shfl_xor(mx, o, 32));
  if ((tid & 31) == 0) red[tid >> 5] = mx;
  __syncthreads();
  mx = red[0];
#pragma unroll
  for (int i = 1; i < 8; ++i) mx = fmaxf(mx, red[i]);
  __syncthreads();

  float sm = 0.f;
  for (int c = tid; c < cols; c += 256) sm += __expf(r[c] - mx);
  for (int o = 16; o > 0; o >>= 1) sm += __shfl_xor(sm, o, 32);
  if ((tid & 31) == 0) red[tid >> 5] = sm;
  __syncthreads();
  sm = 0.f;
#pragma unroll
  for (int i = 0; i < 8; ++i) sm += red[i];
  const float inv = 1.0f / sm;

  for (int c = tid; c < cols; c += 256) {
    const float e = __expf(r[c] - mx) * inv;
    outH[(size_t)row * orow + (size_t)c * ocol] = (f16)e;
  }
}

// --- f32 -> f16 convert -----------------------------------------------------
__global__ __launch_bounds__(256) void cvt_f32_f16_kernel(
    const float* __restrict__ src, f16* __restrict__ dst, int n) {
  const int i = (int)blockIdx.x * 256 + (int)threadIdx.x;
  if (i < n) dst[i] = (f16)src[i];
}

// --- f32 -> f16 convert + transpose: dst[c][r] = src[r][c] ------------------
// rows, cols multiples of 32. Block (32,8); LDS 32x33 to dodge bank conflicts.
__global__ __launch_bounds__(256) void transpose_cvt_kernel(
    const float* __restrict__ src, f16* __restrict__ dst, int rows, int cols) {
  __shared__ f16 tile[32][33];
  const int c0 = (int)blockIdx.x * 32;
  const int r0 = (int)blockIdx.y * 32;
  const int tx = (int)threadIdx.x, ty = (int)threadIdx.y;
#pragma unroll
  for (int i = 0; i < 4; ++i) {
    const int r = r0 + ty + i * 8;
    tile[ty + i * 8][tx] = (f16)src[(size_t)r * cols + c0 + tx];
  }
  __syncthreads();
#pragma unroll
  for (int i = 0; i < 4; ++i) {
    const int c = c0 + ty + i * 8;   // output row (original column)
    dst[(size_t)c * rows + r0 + tx] = tile[tx][ty + i * 8];
  }
}

// ---------------------------------------------------------------------------
extern "C" void kernel_launch(void* const* d_in, const int* in_sizes, int n_in,
                              void* d_out, int out_size, void* d_ws, size_t ws_size,
                              hipStream_t stream) {
  (void)in_sizes; (void)n_in; (void)out_size; (void)ws_size;
  const int N = TOKENS, D = EMBED, H = NHEADS, HD = HEADDIM;

  const float* X  = (const float*)d_in[0];
  const float* Wq = (const float*)d_in[1];
  const float* bq = (const float*)d_in[2];
  const float* Wk = (const float*)d_in[3];
  const float* bk = (const float*)d_in[4];
  const float* Wv = (const float*)d_in[5];
  const float* bv = (const float*)d_in[6];
  const float* Wo = (const float*)d_in[7];
  const float* bo = (const float*)d_in[8];

  // workspace carve (~76 MB)
  char* w = (char*)d_ws;
  auto carve = [&](size_t bytes) -> void* {
    void* p = (void*)w;
    w += (bytes + 255) & ~(size_t)255;
    return p;
  };
  f16*   Xb    = (f16*)carve((size_t)N * D * 2);
  f16*   Wqt   = (f16*)carve((size_t)D * D * 2);        // W^T [out][in]
  f16*   Wkt   = (f16*)carve((size_t)D * D * 2);
  f16*   Wvt   = (f16*)carve((size_t)D * D * 2);
  f16*   Wot   = (f16*)carve((size_t)D * D * 2);
  f16*   Qh    = (f16*)carve((size_t)H * N * HD * 2);   // [H][N][hd]
  f16*   Kh    = (f16*)carve((size_t)H * N * HD * 2);   // [H][N][hd]
  f16*   Vt    = (f16*)carve((size_t)H * N * HD * 2);   // [H][hd][N]  (V^T)
  f16*   attnb = (f16*)carve((size_t)N * D * 2);        // merged heads [N][D]
  float* S     = (float*)carve((size_t)N * N * 4);      // per-head scores
  f16*   p0b   = (f16*)carve((size_t)N * N * 2);        // softmax(QK^T), row-major
  f16*   Pt    = (f16*)carve((size_t)N * N * 2);        // softmax(KK^T)^T
  f16*   pc    = (f16*)carve((size_t)N * N * 2);        // p after step 1
  f16*   pfb   = (f16*)carve((size_t)N * N * 2);        // final p, f16 copy

  float* out_f = (float*)d_out;                          // [N][D]
  float* p_out = out_f + (size_t)N * D;                  // [H][N][N]

  // 1) convert inputs to f16 (weights transposed for the Bt fragment path)
  cvt_f32_f16_kernel<<<(N * D) / 256, 256, 0, stream>>>(X, Xb, N * D);
  dim3 tgrid(D / 32, D / 32), tblk(32, 8);
  transpose_cvt_kernel<<<tgrid, tblk, 0, stream>>>(Wq, Wqt, D, D);
  transpose_cvt_kernel<<<tgrid, tblk, 0, stream>>>(Wk, Wkt, D, D);
  transpose_cvt_kernel<<<tgrid, tblk, 0, stream>>>(Wv, Wvt, D, D);
  transpose_cvt_kernel<<<tgrid, tblk, 0, stream>>>(Wo, Wot, D, D);

  // 2) Q/K/V projections (Q,K head-split; V head-split transposed)
  dim3 gproj(N / 128, D / 128);
  gemm_wmma_kernel<4, true, false, false, true, 1>
      <<<gproj, 256, 0, stream>>>(Xb, D, Wqt, D, bq, nullptr, 0,
                                  1.0f, 0.0f, nullptr, 0, Qh, 0, D, N);
  gemm_wmma_kernel<4, true, false, false, true, 1>
      <<<gproj, 256, 0, stream>>>(Xb, D, Wkt, D, bk, nullptr, 0,
                                  1.0f, 0.0f, nullptr, 0, Kh, 0, D, N);
  gemm_wmma_kernel<4, true, false, false, true, 2>
      <<<gproj, 256, 0, stream>>>(Xb, D, Wvt, D, bv, nullptr, 0,
                                  1.0f, 0.0f, nullptr, 0, Vt, 0, D, N);

  // 3) per-head attention + diffusion (working set ~25 MB -> L2 resident)
  dim3 gNN(N / 128, N / 128);
  dim3 gNV(N / 128, HD / 64);
  const float inv_sqrt_hd = 0.125f;    // 1/sqrt(64)
  const float inv_sqrt_d  = 0.03125f;  // 1/sqrt(1024)

  for (int hidx = 0; hidx < H; ++hidx) {
    const f16* Qp  = Qh + (size_t)hidx * N * HD;
    const f16* Kp  = Kh + (size_t)hidx * N * HD;
    const f16* Vtp = Vt + (size_t)hidx * N * HD;   // [hd][N]
    float* p_out_h = p_out + (size_t)hidx * N * N;

    // scores = Q K^T / sqrt(hd); softmax -> p0 (row-major)
    gemm_wmma_kernel<4, false, false, true, false, 0>
        <<<gNN, 256, 0, stream>>>(Qp, HD, Kp, HD, nullptr, nullptr, 0,
                                  inv_sqrt_hd, 0.0f, S, N, nullptr, 0, HD, 0);
    softmax_rows_kernel<<<N, 256, 0, stream>>>(S, p0b, N, N, 1);

    // sim = K K^T / sqrt(D); softmax -> P, written transposed (P^T)
    gemm_wmma_kernel<4, false, false, true, false, 0>
        <<<gNN, 256, 0, stream>>>(Kp, HD, Kp, HD, nullptr, nullptr, 0,
                                  inv_sqrt_d, 0.0f, S, N, nullptr, 0, HD, 0);
    softmax_rows_kernel<<<N, 256, 0, stream>>>(S, Pt, N, 1, N);

    // diffusion step 1: pc = 0.95*p0 + 0.05*(p0 @ P)
    gemm_wmma_kernel<4, false, true, false, true, 0>
        <<<gNN, 256, 0, stream>>>(p0b, N, Pt, N, nullptr, p0b, N,
                                  ALPHA_C, 1.0f - ALPHA_C,
                                  nullptr, 0, pc, N, N, 0);
    // diffusion step 2: p = 0.95*p0 + 0.05*(pc @ P) -> f32 to d_out + f16 copy
    gemm_wmma_kernel<4, false, true, true, true, 0>
        <<<gNN, 256, 0, stream>>>(pc, N, Pt, N, nullptr, p0b, N,
                                  ALPHA_C, 1.0f - ALPHA_C,
                                  p_out_h, N, pfb, N, N, 0);

    // attn_h = p @ V  (B = V^T[hd][N]) -> merged-head f16 [N][D] at col h*64
    gemm_wmma_kernel<2, false, false, false, true, 0>
        <<<gNV, 256, 0, stream>>>(pfb, N, Vtp, N, nullptr, nullptr, 0,
                                  1.0f, 0.0f, nullptr, 0,
                                  attnb + (size_t)hidx * HD, D, N, 0);
  }

  // 4) output projection: out = attn @ Wo + bo (f32 to d_out)
  dim3 gout(N / 128, D / 128);
  gemm_wmma_kernel<4, true, false, true, false, 0>
      <<<gout, 256, 0, stream>>>(attnb, D, Wot, D, bo, nullptr, 0,
                                 1.0f, 0.0f, out_f, D, nullptr, 0, D, 0);
}
